// MLA_MQA_32255204393141
// MI455X (gfx1250) — matrix operand order repaired and checked by reading, not compiled, not run on hardware
//
#include <hip/hip_runtime.h>

#define D_MODEL   2048
#define NUM_HEADS 16
#define D_K       128
#define SEQ_LEN   4096
#define BATCH     4
#define N_TOTAL   (NUM_HEADS * D_K + 2 * D_K)   // 2304
#define M_TOTAL   (BATCH * SEQ_LEN)             // 16384

constexpr int BM = 128, BN = 128, BK = 32;
constexpr int APITCH = 40;   // fallback: ushorts per A row
constexpr int BPITCH = 40;   // fallback: ushorts per B row
constexpr int CPITCH = 132;  // floats per C row

// TDM kernel tiling: BK2=64, TDM pads 4 dwords every 16 dwords stored
constexpr int BK2    = 64;
constexpr int PITCH2 = 80;   // ushorts per row (32dw data + 2x4dw pad)
constexpr int SUBOFF = 40;   // ushort offset of k-subpanel 1 (20 dwords)
constexpr int NK2    = D_MODEL / BK2;            // 32
constexpr int ABUF_B = BM * PITCH2 * 2;          // 20480 bytes per A buffer

typedef __attribute__((ext_vector_type(16))) __bf16        v16bf;
typedef __attribute__((ext_vector_type(8)))  __bf16        v8bf;
typedef __attribute__((ext_vector_type(8)))  float         v8f;
typedef __attribute__((ext_vector_type(4)))  unsigned int  v4u;
typedef __attribute__((ext_vector_type(8)))  int           v8i;
typedef __attribute__((ext_vector_type(4)))  int           v4i;

__device__ __forceinline__ unsigned short f2bf(float f) {
  union { float f; unsigned u; } v; v.f = f;
  unsigned u = v.u;
  return (unsigned short)((u + 0x7FFFu + ((u >> 16) & 1u)) >> 16);  // RNE
}
__device__ __forceinline__ unsigned pack2bf(float lo, float hi) {
  return (unsigned)f2bf(lo) | ((unsigned)f2bf(hi) << 16);
}

// ---------------------------------------------------------------------------
// Pack kernels: fp32 -> bf16 workspace (x row-major, W transposed [n][k])
// ---------------------------------------------------------------------------
__global__ __launch_bounds__(256)
void pack_x(const float* __restrict__ x, unsigned* __restrict__ xb) {
  const long i = ((long)blockIdx.x * 256 + threadIdx.x) * 4;
  const float4 v = *reinterpret_cast<const float4*>(x + i);
  uint2 o; o.x = pack2bf(v.x, v.y); o.y = pack2bf(v.z, v.w);
  *reinterpret_cast<uint2*>(xb + (i >> 1)) = o;
}

__global__ __launch_bounds__(256)
void pack_w(const float* __restrict__ Wq, const float* __restrict__ Wk,
            const float* __restrict__ Wv, unsigned short* __restrict__ wt) {
  __shared__ float t[32][33];
  const int k0  = blockIdx.x * 32;
  const int n0g = blockIdx.y * 32;
  const float* W; int ldW, col0;
  if (n0g < NUM_HEADS * D_K)            { W = Wq; ldW = NUM_HEADS * D_K; col0 = n0g; }
  else if (n0g < NUM_HEADS * D_K + D_K) { W = Wk; ldW = D_K; col0 = n0g - NUM_HEADS * D_K; }
  else                                  { W = Wv; ldW = D_K; col0 = n0g - NUM_HEADS * D_K - D_K; }
  const int c = threadIdx.x & 31, r4 = threadIdx.x >> 5;
  #pragma unroll
  for (int p = 0; p < 4; ++p) {
    const int r = r4 + p * 8;
    t[r][c] = W[(long)(k0 + r) * ldW + col0 + c];
  }
  __syncthreads();
  #pragma unroll
  for (int p = 0; p < 4; ++p) {
    const int n = r4 + p * 8;
    wt[(long)(n0g + n) * D_MODEL + k0 + c] = f2bf(t[c][n]);
  }
}

// ---------------------------------------------------------------------------
// TDM descriptor build + issue (ISA 08_async_tensor §8): 2-D tile tileK x 128,
// 2-byte elements, tensor row stride D_MODEL, pad 4 dwords every 16 dwords
// (=> PITCH2-ushort LDS pitch matching WMMA fragment addressing).
// ---------------------------------------------------------------------------
__device__ __forceinline__ void tdm_load_tile(unsigned ldsOff,
                                              const unsigned short* gptr,
                                              unsigned tensorDim1,
                                              unsigned tileK) {
  const unsigned long long ga = (unsigned long long)(uintptr_t)gptr;
  v4u g0;
  g0[0] = 1u;                                        // count=1, user mode
  g0[1] = ldsOff;                                    // lds_addr
  g0[2] = (unsigned)(ga & 0xFFFFFFFFu);              // global_addr[31:0]
  g0[3] = (unsigned)((ga >> 32) & 0x01FFFFFFu) | (2u << 30);  // addr[56:32]|type=2
  v8i g1;
  g1[0] = (int)(0x10000u        // data_size = 1 (2 bytes)
              | 0x100000u       // pad_enable
              | (3u << 22)      // pad_interval: every 16 dwords
              | (3u << 25));    // pad_amount: 4 dwords
  g1[1] = (int)((unsigned)(D_MODEL & 0xFFFF) << 16);            // tensor_dim0 lo
  g1[2] = (int)(((unsigned)D_MODEL >> 16) | ((tensorDim1 & 0xFFFFu) << 16));
  g1[3] = (int)((tensorDim1 >> 16) | (tileK << 16));            // tile_dim0
  g1[4] = (int)(unsigned)BN;                                    // tile_dim1 = 128
  g1[5] = (int)D_MODEL;                                         // dim0_stride lo32
  g1[6] = (int)((unsigned)(D_MODEL & 0xFFFF) << 16);            // dim1_stride lo16
  g1[7] = 0;
  const v4i z4 = {0, 0, 0, 0};
#if __has_include(<hip/amd_detail/amd_gfx1250_TDM.h>)
  const v8i z8 = {0, 0, 0, 0, 0, 0, 0, 0};
  __builtin_amdgcn_tensor_load_to_lds(g0, g1, z4, z4, z8, 0);
#else
  __builtin_amdgcn_tensor_load_to_lds(g0, g1, z4, z4, 0);
#endif
}

// ---------------------------------------------------------------------------
// Shared epilogue: bias + sqrt(dk) scale + interleaved-RoPE + scatter
// ---------------------------------------------------------------------------
__device__ __forceinline__ void rope_epilogue(const float* sC, const float* bias,
                                              int tn, long mBase, int tid,
                                              float* __restrict__ out) {
  const int  jo     = tid & 127;
  const int  j      = jo & 63;
  const bool hiHalf = jo >= 64;
  const bool useSin = j < 32;
  const int  t1     = useSin ? 2 * j : 2 * j - 64;
  const float kLn   = 9.210340371976184f / 64.0f;      // ln(10000)/64
  const float th1   = __expf(-kLn * (float)t1);
  const float th2   = __expf(-kLn * (float)(t1 + 1));
  const int  c1 = 2 * j, c2 = 2 * j + 1;
  const float b1 = bias[c1], b2 = bias[c2];
  const float scale = 11.313708498984761f;             // sqrt(128)
  const long QTOT = (long)BATCH * NUM_HEADS * SEQ_LEN * D_K;
  const long KTOT = (long)BATCH * SEQ_LEN * D_K;

  for (int mm = (tid >> 7); mm < BM; mm += 2) {
    const long mg   = mBase + mm;
    const int  bidx = (int)(mg >> 12);
    const int  s    = (int)(mg & (SEQ_LEN - 1));
    const float r1 = useSin ? __sinf((float)s * th1) : __cosf((float)s * th1);
    const float r2 = useSin ? __sinf((float)s * th2) : __cosf((float)s * th2);
    const float v1 = (sC[mm * CPITCH + c1] + b1) * scale;
    const float v2 = (sC[mm * CPITCH + c2] + b2) * scale;
    const float o  = hiHalf ? (v2 * r1 + v1 * r2) : (v1 * r1 - v2 * r2);
    long off;
    if (tn < NUM_HEADS)       off = (((long)bidx * NUM_HEADS + tn) * SEQ_LEN + s) * D_K + jo;
    else if (tn == NUM_HEADS) off = QTOT + ((long)bidx * SEQ_LEN + s) * D_K + jo;
    else                      off = QTOT + KTOT + ((long)bidx * SEQ_LEN + s) * D_K + jo;
    out[off] = o;
  }
}

// ---------------------------------------------------------------------------
// Fast path: bf16 workspace + double-buffered TDM staging + WMMA
// LDS: A[2]+B[2] staging (80 KB) overlaid with the 67.5 KB epilogue tile.
// ---------------------------------------------------------------------------
__global__ __launch_bounds__(256)
void qkv_rope_wmma_tdm(const unsigned short* __restrict__ xb,   // [M][2048] bf16
                       const unsigned short* __restrict__ wt,   // [2304][2048] bf16
                       const float* __restrict__ bq, const float* __restrict__ bk,
                       const float* __restrict__ bv, float* __restrict__ out) {
  __shared__ __align__(16) char smem[4 * ABUF_B];               // 81920 B
  unsigned short* sA = reinterpret_cast<unsigned short*>(smem);            // [2][BM*PITCH2]
  unsigned short* sB = reinterpret_cast<unsigned short*>(smem + 2 * ABUF_B);
  float*          sC = reinterpret_cast<float*>(smem);          // overlay (post-loop)

  const int tn   = blockIdx.x;
  const int tm   = blockIdx.y;
  const int tid  = threadIdx.x;
  const int lane = tid & 31;
  const int wave = tid >> 5;
  const int wm   = wave >> 1;
  const int wn   = wave & 1;

  const float* bias = (tn < NUM_HEADS) ? (bq + tn * D_K)
                     : (tn == NUM_HEADS) ? bk : bv;
  const long mBase = (long)tm * BM;
  const long nBase = (long)tn * BN;

  const unsigned sAoff = (unsigned)(unsigned long long)(uintptr_t)sA;
  const unsigned sBoff = (unsigned)(unsigned long long)(uintptr_t)sB;

  v8f acc[2][4];
  #pragma unroll
  for (int i = 0; i < 2; ++i)
    #pragma unroll
    for (int j = 0; j < 4; ++j)
      #pragma unroll
      for (int e = 0; e < 8; ++e) acc[i][j][e] = 0.0f;

  const int half = lane >> 4;
  const int l16  = lane & 15;
  const int kbA  = half * 16;
  const int khB  = half * 16;

  // prologue: prefetch tile 0 into buffer 0
  if (wave == 0) {
    tdm_load_tile(sAoff, xb + mBase * D_MODEL, (unsigned)M_TOTAL, (unsigned)BK2);
    tdm_load_tile(sBoff, wt + nBase * D_MODEL, (unsigned)N_TOTAL, (unsigned)BK2);
  }

  for (int kt = 0; kt < NK2; ++kt) {
    const int cur = kt & 1;
    if (wave == 0) __builtin_amdgcn_s_wait_tensorcnt(0);   // tile kt landed
    __syncthreads();   // publish LDS; all waves done reading the other buffer
    if (wave == 0 && kt + 1 < NK2) {                       // prefetch tile kt+1
      const int nxt = cur ^ 1;
      const int k1  = (kt + 1) * BK2;
      tdm_load_tile(sAoff + nxt * ABUF_B, xb + mBase * D_MODEL + k1,
                    (unsigned)M_TOTAL, (unsigned)BK2);
      tdm_load_tile(sBoff + nxt * ABUF_B, wt + nBase * D_MODEL + k1,
                    (unsigned)N_TOTAL, (unsigned)BK2);
    }

    const unsigned short* pA = sA + cur * (BM * PITCH2);
    const unsigned short* pB = sB + cur * (BN * PITCH2);
    #pragma unroll
    for (int s = 0; s < 2; ++s) {          // two WMMA k-steps per staged tile
      union FA { v16bf v; v8bf h[2]; } a[2];
      #pragma unroll
      for (int i = 0; i < 2; ++i) {
        const unsigned short* base =
            pA + (wm * 32 + i * 16 + l16) * PITCH2 + s * SUBOFF;
        a[i].h[0] = *reinterpret_cast<const v8bf*>(base + kbA);
        a[i].h[1] = *reinterpret_cast<const v8bf*>(base + kbA + 8);
      }
      union FB { v16bf v; v8bf h[2]; } bf[4];
      #pragma unroll
      for (int j = 0; j < 4; ++j) {
        const unsigned short* base =
            pB + (wn * 64 + j * 16 + l16) * PITCH2 + s * SUBOFF + khB;
        bf[j].h[0] = *reinterpret_cast<const v8bf*>(base);
        bf[j].h[1] = *reinterpret_cast<const v8bf*>(base + 8);
      }
      #pragma unroll
      for (int i = 0; i < 2; ++i)
        #pragma unroll
        for (int j = 0; j < 4; ++j)
          acc[i][j] = __builtin_amdgcn_wmma_f32_16x16x32_bf16(
              false, a[i].v, false, bf[j].v, (short)0, acc[i][j], false, false);
    }
  }

  __syncthreads();   // staging buffers dead; safe to overlay sC
  {
    const int mh = (lane >> 4) * 8;
    #pragma unroll
    for (int i = 0; i < 2; ++i)
      #pragma unroll
      for (int j = 0; j < 4; ++j) {
        float* c = &sC[(wm * 32 + i * 16 + mh) * CPITCH + wn * 64 + j * 16 + l16];
        #pragma unroll
        for (int r = 0; r < 8; ++r) c[r * CPITCH] = acc[i][j][r];
      }
  }
  __syncthreads();
  rope_epilogue(sC, bias, tn, mBase, tid, out);
}

// ---------------------------------------------------------------------------
// Fallback path (no workspace): fused fp32->bf16 staging + WMMA
// ---------------------------------------------------------------------------
__global__ __launch_bounds__(256)
void qkv_rope_wmma(const float* __restrict__ x,
                   const float* __restrict__ Wq, const float* __restrict__ bq,
                   const float* __restrict__ Wk, const float* __restrict__ bk,
                   const float* __restrict__ Wv, const float* __restrict__ bv,
                   float* __restrict__ out) {
  __shared__ __align__(16) unsigned short sA[BM * APITCH];
  __shared__ __align__(16) unsigned short sB[BN * BPITCH];
  __shared__ __align__(16) float          sC[BM * CPITCH];

  const int tn   = blockIdx.x;
  const int tm   = blockIdx.y;
  const int tid  = threadIdx.x;
  const int lane = tid & 31;
  const int wave = tid >> 5;
  const int wm   = wave >> 1;
  const int wn   = wave & 1;

  const float* W; const float* bias; int ldW, cb;
  if (tn < NUM_HEADS)      { W = Wq; ldW = NUM_HEADS * D_K; cb = tn * D_K; bias = bq + cb; }
  else if (tn == NUM_HEADS){ W = Wk; ldW = D_K;             cb = 0;        bias = bk; }
  else                     { W = Wv; ldW = D_K;             cb = 0;        bias = bv; }

  const long mBase = (long)tm * BM;

  v8f acc[2][4];
  #pragma unroll
  for (int i = 0; i < 2; ++i)
    #pragma unroll
    for (int j = 0; j < 4; ++j)
      #pragma unroll
      for (int e = 0; e < 8; ++e) acc[i][j][e] = 0.0f;

  const int arow = tid >> 3;
  const int acol = (tid & 7) * 4;
  const int brow = tid >> 5;
  const int bcol = (tid & 31) * 4;

  const int half = lane >> 4;
  const int l16  = lane & 15;
  const int kbA  = half * 16;
  const int khB  = half * 16;

  for (int kt = 0; kt < D_MODEL / BK; ++kt) {
    const int k0 = kt * BK;
    __syncthreads();
    #pragma unroll
    for (int p = 0; p < 4; ++p) {
      const int r = arow + p * 32;
      const float4 v = *reinterpret_cast<const float4*>(
          &x[(mBase + r) * D_MODEL + k0 + acol]);
      unsigned* d = reinterpret_cast<unsigned*>(&sA[r * APITCH + acol]);
      d[0] = pack2bf(v.x, v.y); d[1] = pack2bf(v.z, v.w);
    }
    #pragma unroll
    for (int p = 0; p < 4; ++p) {
      const int r = brow + p * 8;
      const float4 v = *reinterpret_cast<const float4*>(
          &W[(long)(k0 + r) * ldW + cb + bcol]);
      sB[(bcol + 0) * BPITCH + r] = f2bf(v.x);
      sB[(bcol + 1) * BPITCH + r] = f2bf(v.y);
      sB[(bcol + 2) * BPITCH + r] = f2bf(v.z);
      sB[(bcol + 3) * BPITCH + r] = f2bf(v.w);
    }
    __syncthreads();

    union FA { v16bf v; v8bf h[2]; } a[2];
    #pragma unroll
    for (int i = 0; i < 2; ++i) {
      const unsigned short* base = &sA[(wm * 32 + i * 16 + l16) * APITCH];
      a[i].h[0] = *reinterpret_cast<const v8bf*>(base + kbA);
      a[i].h[1] = *reinterpret_cast<const v8bf*>(base + kbA + 8);
    }
    union FB { v16bf v; v8bf h[2]; } bf[4];
    #pragma unroll
    for (int j = 0; j < 4; ++j) {
      const unsigned short* base = &sB[(wn * 64 + j * 16 + l16) * BPITCH + khB];
      bf[j].h[0] = *reinterpret_cast<const v8bf*>(base);
      bf[j].h[1] = *reinterpret_cast<const v8bf*>(base + 8);
    }
    #pragma unroll
    for (int i = 0; i < 2; ++i)
      #pragma unroll
      for (int j = 0; j < 4; ++j)
        acc[i][j] = __builtin_amdgcn_wmma_f32_16x16x32_bf16(
            false, a[i].v, false, bf[j].v, (short)0, acc[i][j], false, false);
  }

  __syncthreads();
  {
    const int mh = (lane >> 4) * 8;
    #pragma unroll
    for (int i = 0; i < 2; ++i)
      #pragma unroll
      for (int j = 0; j < 4; ++j) {
        float* c = &sC[(wm * 32 + i * 16 + mh) * CPITCH + wn * 64 + j * 16 + l16];
        #pragma unroll
        for (int r = 0; r < 8; ++r) c[r * CPITCH] = acc[i][j][r];
      }
  }
  __syncthreads();
  rope_epilogue(sC, bias, tn, mBase, tid, out);
}

// ---------------------------------------------------------------------------
extern "C" void kernel_launch(void* const* d_in, const int* in_sizes, int n_in,
                              void* d_out, int out_size, void* d_ws, size_t ws_size,
                              hipStream_t stream) {
  (void)in_sizes; (void)n_in; (void)out_size;
  const float* x  = (const float*)d_in[0];
  const float* Wq = (const float*)d_in[1];
  const float* bq = (const float*)d_in[2];
  const float* Wk = (const float*)d_in[3];
  const float* bk = (const float*)d_in[4];
  const float* Wv = (const float*)d_in[5];
  const float* bv = (const float*)d_in[6];

  const size_t XB_BYTES = (size_t)M_TOTAL * D_MODEL * 2;   // 67 MB
  const size_t WT_BYTES = (size_t)N_TOTAL * D_MODEL * 2;   // 9.4 MB
  dim3 grid(NUM_HEADS + 2, M_TOTAL / BM);                  // 18 x 128

  if (ws_size >= XB_BYTES + WT_BYTES) {
    unsigned*       xb  = (unsigned*)d_ws;
    unsigned short* wtp = (unsigned short*)((char*)d_ws + XB_BYTES);
    pack_x<<<((long)M_TOTAL * D_MODEL) / (256 * 4), 256, 0, stream>>>(x, xb);
    dim3 gw(D_MODEL / 32, N_TOTAL / 32);
    pack_w<<<gw, 256, 0, stream>>>(Wq, Wk, Wv, wtp);
    qkv_rope_wmma_tdm<<<grid, 256, 0, stream>>>((const unsigned short*)xb, wtp,
                                                bq, bk, bv, (float*)d_out);
  } else {
    qkv_rope_wmma<<<grid, 256, 0, stream>>>(x, Wq, bq, Wk, bk, Wv, bv,
                                            (float*)d_out);
  }
}
